// GraphConvolution_12781822673573
// MI455X (gfx1250) — matrix operand order repaired and checked by reading, not compiled, not run on hardware
//
#include <hip/hip_runtime.h>

typedef _Float16 v16h __attribute__((ext_vector_type(16)));
typedef _Float16 v4h  __attribute__((ext_vector_type(4)));
typedef float    v8f  __attribute__((ext_vector_type(8)));

#define NN     4096            // rows == cols == 4096
#define KSPLIT 8
#define KRANGE (NN / KSPLIT)   // 512
#define NCHUNK (KRANGE / 32)   // 16
#define ASTRIDE 36             // halfs per A row  (72B: 8B-aligned quads, conflict-free gather)
#define BSTRIDE 36             // halfs per B row
#define LDS_AS  (16 * ASTRIDE) // 576
#define LDS_BS  (64 * BSTRIDE) // 2304
#define LDS_WAVE (LDS_AS + LDS_BS)

// ---------------- WMMA fragment helpers (f16 16x16x32, wave32 layouts) --------

// Stage B chunk: Bs[n*BSTRIDE + k] = (half) Bsrc[(k0+k)*64 + n], k in [0,32), n in [0,64)
// b128 loads: lanes 0-15 cover one k-row of 64 floats.
__device__ __forceinline__ void stage_B_f16(const float* __restrict__ Bsrc, int k0,
                                            _Float16* Bs, int lane) {
  #pragma unroll 4
  for (int i = 0; i < 16; ++i) {
    int idx = i * 32 + lane;          // 512 float4 total
    int k  = idx >> 4;                // 0..31
    int n4 = (idx & 15) * 4;          // 0,4,..,60
    float4 v = *(const float4*)&Bsrc[(size_t)(k0 + k) * 64 + n4];
    Bs[(n4 + 0) * BSTRIDE + k] = (_Float16)v.x;
    Bs[(n4 + 1) * BSTRIDE + k] = (_Float16)v.y;
    Bs[(n4 + 2) * BSTRIDE + k] = (_Float16)v.z;
    Bs[(n4 + 3) * BSTRIDE + k] = (_Float16)v.w;
  }
}

// Stage plain A chunk (dense layers): As[r*ASTRIDE + kl] = (half)Asrc[(r0+r)*64 + k0 + kl]
__device__ __forceinline__ void stage_A_f16(const float* __restrict__ Asrc, int r0, int k0,
                                            _Float16* As, int lane) {
  int quad = lane & 7;   // 8 quads of 4 cols
  int rg   = lane >> 3;  // 4 row-groups
  #pragma unroll
  for (int rr = 0; rr < 4; ++rr) {
    int r = rr * 4 + rg;
    float4 v = *(const float4*)&Asrc[(size_t)(r0 + r) * 64 + k0 + quad * 4];
    v4h h;
    h[0] = (_Float16)v.x; h[1] = (_Float16)v.y;
    h[2] = (_Float16)v.z; h[3] = (_Float16)v.w;
    *(v4h*)(As + r * ASTRIDE + quad * 4) = h;
  }
}

// A fragment (16x32 f16): lane<16 -> M=lane, K in {0..15}; lane>=16 -> K in {16..31}
__device__ __forceinline__ v16h gather_A(const _Float16* As, int lane) {
  int row = lane & 15;
  int kb  = (lane >> 4) * 16;
  const _Float16* p = As + row * ASTRIDE;
  v16h a;
  #pragma unroll
  for (int j = 0; j < 8; ++j) {
    int k = kb + 2 * j;
    a[2 * j]     = p[k];
    a[2 * j + 1] = p[k + 1];
  }
  return a;
}

// B fragment (32x16 f16) for N-tile t: lanes 0-15 hold K=0..15 (VGPR j = K 2j,2j+1), lanes 16-31 K=16..31
__device__ __forceinline__ v16h gather_B(const _Float16* Bs, int t, int lane) {
  int n  = t * 16 + (lane & 15);
  int kb = (lane >> 4) * 16;
  const _Float16* p = Bs + n * BSTRIDE + kb;
  v16h b;
  #pragma unroll
  for (int j = 0; j < 8; ++j) { b[2 * j] = p[2 * j]; b[2 * j + 1] = p[2 * j + 1]; }
  return b;
}

// ---------------- small kernels ----------------------------------------------

// out[i] = dot(X[i, 0:64], w)
__global__ __launch_bounds__(256) void dot64_kernel(const float* __restrict__ X,
                                                    const float* __restrict__ w,
                                                    float* __restrict__ out, int n) {
  int i = blockIdx.x * blockDim.x + threadIdx.x;
  if (i >= n) return;
  const float4* x4 = (const float4*)(X + (size_t)i * 64);
  const float4* w4 = (const float4*)w;
  float s = 0.f;
  #pragma unroll
  for (int j = 0; j < 16; ++j) {
    float4 a = x4[j], b = w4[j];
    s += a.x * b.x + a.y * b.y + a.z * b.z + a.w * b.w;
  }
  out[i] = s;
}

// node->obj attention + obj update + bias precompute for the next dense layer.
__global__ __launch_bounds__(256) void obj_update_kernel(
    const float* __restrict__ hid,  const float* __restrict__ supp,
    const float* __restrict__ prev, const float* __restrict__ w_attn,
    const float* __restrict__ b_attn, const float* __restrict__ W_up,
    const float* __restrict__ W_top, float* __restrict__ obj_out,
    float* __restrict__ bias_out) {
  __shared__ float s_a[NN];
  __shared__ float s_red[256];
  __shared__ float s_vec[64];
  __shared__ float s_obj[64];
  int t = threadIdx.x;
  float oc = b_attn[0];
  #pragma unroll 8
  for (int k = 0; k < 64; ++k) oc += prev[k] * w_attn[66 + k];
  for (int c = t; c < NN; c += 256) {
    const float* hp = hid + (size_t)c * 64;
    float d = 0.f;
    #pragma unroll 8
    for (int k = 0; k < 64; ++k) d += hp[k] * w_attn[k];
    d += supp[c * 2] * w_attn[64] + supp[c * 2 + 1] * w_attn[65] + oc;
    s_a[c] = 1.0f / (1.0f + __expf(-d));
  }
  __syncthreads();
  int h = t & 63, part = t >> 6;
  float acc = 0.f;
  for (int c = part * 1024; c < part * 1024 + 1024; ++c)
    acc += s_a[c] * hid[(size_t)c * 64 + h];
  s_red[t] = acc;
  __syncthreads();
  if (t < 64) s_vec[t] = s_red[t] + s_red[t + 64] + s_red[t + 128] + s_red[t + 192];
  __syncthreads();
  if (t < 64) {
    float a2 = 0.f;
    #pragma unroll 8
    for (int k = 0; k < 64; ++k)
      a2 += prev[k] * W_up[k * 64 + t] + s_vec[k] * W_up[(64 + k) * 64 + t];
    float o = fmaxf(a2, 0.f);
    s_obj[t] = o;
    obj_out[t] = o;
  }
  __syncthreads();
  if (t < 64) {
    float b2 = 0.f;
    #pragma unroll 8
    for (int k = 0; k < 64; ++k) b2 += s_obj[k] * W_top[k * 64 + t];
    bias_out[t] = b2;
  }
}

// out = relu(bias + A1 @ W1 [+ A2 @ W2]) ; A*: [4096][64] f32, W*: [64][64] f32.
// 8 independent waves/block (per-wave LDS slice), 16-row WMMA tile per wave.
__global__ __launch_bounds__(256) void dense_relu_kernel(
    const float* __restrict__ A1, const float* __restrict__ W1,
    const float* __restrict__ A2, const float* __restrict__ W2,
    const float* __restrict__ bias, float* __restrict__ out) {
  __shared__ _Float16 lds[8][LDS_WAVE];
  int wave = threadIdx.x >> 5, lane = threadIdx.x & 31;
  _Float16* As = lds[wave];
  _Float16* Bs = lds[wave] + LDS_AS;
  int r0 = (blockIdx.x * 8 + wave) * 16;
  v8f acc[4] = {v8f{}, v8f{}, v8f{}, v8f{}};
  int npass = (A2 != nullptr) ? 2 : 1;
  for (int pass = 0; pass < npass; ++pass) {
    const float* Asrc = pass ? A2 : A1;
    const float* Wsrc = pass ? W2 : W1;
    for (int ch = 0; ch < 2; ++ch) {
      int k0 = ch * 32;
      stage_B_f16(Wsrc, k0, Bs, lane);
      stage_A_f16(Asrc, r0, k0, As, lane);
      __builtin_amdgcn_wave_barrier();   // DS in-order per wave; fence compiler only
      v16h a = gather_A(As, lane);
      #pragma unroll
      for (int t = 0; t < 4; ++t) {
        v16h b = gather_B(Bs, t, lane);
        acc[t] = __builtin_amdgcn_wmma_f32_16x16x32_f16(false, a, false, b,
                                                        (short)0, acc[t], false, false);
      }
      __builtin_amdgcn_wave_barrier();
    }
  }
  int mrow = (lane >> 4) * 8;
  int n0 = lane & 15;
  #pragma unroll
  for (int t = 0; t < 4; ++t) {
    int n = t * 16 + n0;
    float bn = bias ? bias[n] : 0.f;
    #pragma unroll
    for (int i = 0; i < 8; ++i)
      out[(size_t)(r0 + mrow + i) * 64 + n] = fmaxf(acc[t][i] + bn, 0.f);
  }
}

// Fused masked-sigmoid attention GEMM (K-split partials):
// score[r,c] = (supp0[r,c]!=0) ? sigmoid(rdot[r]+kdot[c]+supp0*w0+supp1*w1+b) : 0
// partials[ks][r][:] += score[r, ksrange] @ B[ksrange, :64]
// Block = 1 M-tile (16 rows); 8 independent waves = 8 K-splits.
__global__ __launch_bounds__(256) void fused_attn_partial_kernel(
    const float* __restrict__ supp0, const float* __restrict__ supp1,
    const float* __restrict__ rdot,  const float* __restrict__ kdot,
    const float* __restrict__ wsup,  const float* __restrict__ bptr,
    const float* __restrict__ B,     float* __restrict__ partials) {
  __shared__ _Float16 lds[8][LDS_WAVE];
  int wave = threadIdx.x >> 5, lane = threadIdx.x & 31;
  _Float16* As = lds[wave];
  _Float16* Bs = lds[wave] + LDS_AS;
  int r0 = blockIdx.x * 16;
  int ks = wave;
  float ws0 = wsup[0], ws1 = wsup[1], bb = bptr[0];
  int quad = lane & 7;   // 4-column group for b128 supp loads
  int rg   = lane >> 3;  // 4 row-groups
  v8f acc[4] = {v8f{}, v8f{}, v8f{}, v8f{}};
  for (int ch = 0; ch < NCHUNK; ++ch) {
    int k0 = ks * KRANGE + ch * 32;
    stage_B_f16(B, k0, Bs, lane);
    // prefetch next chunk of the supp stream (one instr covers 16 rows x 128B)
    if (ch + 1 < NCHUNK) {
      size_t poff = (size_t)(r0 + (lane & 15)) * NN + (size_t)(k0 + 32) + ((lane >> 4) << 4);
      __builtin_prefetch(supp0 + poff, 0, 3);
      __builtin_prefetch(supp1 + poff, 0, 3);
    }
    // stage masked sigmoid scores as f16 (b128 supp loads: 4 rows per pass)
    float4 kd = *(const float4*)(kdot + k0 + quad * 4);
    #pragma unroll
    for (int rr = 0; rr < 4; ++rr) {
      int r = rr * 4 + rg;
      size_t off = (size_t)(r0 + r) * NN + k0 + quad * 4;
      float4 s0 = *(const float4*)(supp0 + off);
      float4 s1 = *(const float4*)(supp1 + off);
      float base = rdot[r0 + r] + bb;
      float lx = base + kd.x + s0.x * ws0 + s1.x * ws1;
      float ly = base + kd.y + s0.y * ws0 + s1.y * ws1;
      float lz = base + kd.z + s0.z * ws0 + s1.z * ws1;
      float lw = base + kd.w + s0.w * ws0 + s1.w * ws1;
      v4h h;
      h[0] = (_Float16)((s0.x != 0.f) ? (1.0f / (1.0f + __expf(-lx))) : 0.f);
      h[1] = (_Float16)((s0.y != 0.f) ? (1.0f / (1.0f + __expf(-ly))) : 0.f);
      h[2] = (_Float16)((s0.z != 0.f) ? (1.0f / (1.0f + __expf(-lz))) : 0.f);
      h[3] = (_Float16)((s0.w != 0.f) ? (1.0f / (1.0f + __expf(-lw))) : 0.f);
      *(v4h*)(As + r * ASTRIDE + quad * 4) = h;
    }
    __builtin_amdgcn_wave_barrier();
    v16h a = gather_A(As, lane);
    #pragma unroll
    for (int t = 0; t < 4; ++t) {
      v16h b = gather_B(Bs, t, lane);
      acc[t] = __builtin_amdgcn_wmma_f32_16x16x32_f16(false, a, false, b,
                                                      (short)0, acc[t], false, false);
    }
    __builtin_amdgcn_wave_barrier();
  }
  int mrow = (lane >> 4) * 8;
  int n0 = lane & 15;
  float* outp = partials + ((size_t)ks * NN + r0) * 64;
  #pragma unroll
  for (int t = 0; t < 4; ++t)
    #pragma unroll
    for (int i = 0; i < 8; ++i)
      outp[(size_t)(mrow + i) * 64 + t * 16 + n0] = acc[t][i];
}

// deterministic partial reduction: out[i] = sum_ks partials[ks][i]  (float4)
__global__ __launch_bounds__(256) void reduce8_kernel(const float* __restrict__ p,
                                                      float* __restrict__ out) {
  size_t i = ((size_t)blockIdx.x * blockDim.x + threadIdx.x) * 4;  // 65536 threads
  float4 s = {0.f, 0.f, 0.f, 0.f};
  #pragma unroll
  for (int k = 0; k < KSPLIT; ++k) {
    float4 v = *(const float4*)(p + (size_t)k * (NN * 64) + i);
    s.x += v.x; s.y += v.y; s.z += v.z; s.w += v.w;
  }
  *(float4*)(out + i) = s;
}

// ---------------- host orchestration -----------------------------------------

extern "C" void kernel_launch(void* const* d_in, const int* in_sizes, int n_in,
                              void* d_out, int out_size, void* d_ws, size_t ws_size,
                              hipStream_t stream) {
  const float* col_hidden = (const float*)d_in[0];
  const float* row_hidden = (const float*)d_in[1];
  const float* obj_hidden = (const float*)d_in[2];
  const float* cv_supp    = (const float*)d_in[3];   // [2,4096,4096]
  const float* vc_supp    = (const float*)d_in[4];   // [2,4096,4096]
  const float* vo_supp    = (const float*)d_in[5];   // [4096,2]
  const float* co_supp    = (const float*)d_in[6];   // [4096,2]
  const float* W_vo       = (const float*)d_in[7];
  const float* W_oc       = (const float*)d_in[8];
  const float* W_vc       = (const float*)d_in[9];
  const float* W_co       = (const float*)d_in[10];
  const float* W_ov       = (const float*)d_in[11];
  const float* W_cv       = (const float*)d_in[12];
  const float* w_attn_vo  = (const float*)d_in[13];
  const float* b_attn_vo  = (const float*)d_in[14];
  const float* w_attn_co  = (const float*)d_in[15];
  const float* b_attn_co  = (const float*)d_in[16];
  const float* w_cv_col   = (const float*)d_in[17];
  const float* w_cv_supp  = (const float*)d_in[18];
  const float* w_cv_row   = (const float*)d_in[19];
  const float* b_cv       = (const float*)d_in[20];
  const float* w_vc_row   = (const float*)d_in[21];
  const float* w_vc_supp  = (const float*)d_in[22];
  const float* w_vc_col   = (const float*)d_in[23];
  const float* b_vc       = (const float*)d_in[24];

  float* out = (float*)d_out;
  float* ws  = (float*)d_ws;

  // workspace layout (floats)
  float* rdot_cv  = ws;                 // 4096
  float* cdot_cv  = ws + 4096;          // 4096
  float* colpart  = ws + 8192;          // 4096 (col_hidden @ w_vc_col)
  float* rowpart  = ws + 12288;         // 4096 (row_next  @ w_vc_row)
  float* obj1     = ws + 16384;         // 64
  float* bias_oc  = ws + 16448;         // 64
  float* bias_ov  = ws + 16512;         // 64
  float* o_c_out  = ws + 16576;         // 262144
  float* v_out    = o_c_out + 262144;   // 262144
  float* o_v_out  = v_out   + 262144;   // 262144
  float* c_out    = o_v_out + 262144;   // 262144
  float* partials = c_out   + 262144;   // 8*262144 = 2097152

  // output layout (return order, flat)
  const size_t PLANE = (size_t)NN * NN;            // 16777216
  float* out_col = out;                            // 262144
  float* out_row = out + 262144;                   // 262144
  float* out_obj = out + 524288;                   // 64
  float* out_cv  = out + 524352;                   // 2*PLANE
  float* out_vc  = out_cv + 2 * PLANE;             // 2*PLANE
  float* out_vo  = out_vc + 2 * PLANE;             // 8192
  float* out_co  = out_vo + 8192;                  // 8192

  // ---- precompute per-node dot scalars for pairwise logits
  dot64_kernel<<<16, 256, 0, stream>>>(row_hidden, w_cv_row, rdot_cv, NN);
  dot64_kernel<<<16, 256, 0, stream>>>(col_hidden, w_cv_col, cdot_cv, NN);
  dot64_kernel<<<16, 256, 0, stream>>>(col_hidden, w_vc_col, colpart, NN);

  // ---- v -> o : obj1 = relu([obj|v_o_out]@W_vo); bias_oc = obj1 @ W_oc[0:64]
  obj_update_kernel<<<1, 256, 0, stream>>>(col_hidden, vo_supp, obj_hidden,
                                           w_attn_vo, b_attn_vo, W_vo, W_oc,
                                           obj1, bias_oc);

  // ---- o_c_out = relu(bias_oc + row_hidden @ W_oc[64:128])
  dense_relu_kernel<<<32, 256, 0, stream>>>(row_hidden, W_oc + 64 * 64,
                                            nullptr, nullptr, bias_oc, o_c_out);

  // ---- v -> c : v_out = score_cv @ col_hidden  (fused, K-split partials)
  fused_attn_partial_kernel<<<256, 256, 0, stream>>>(cv_supp, cv_supp + PLANE,
                                                     rdot_cv, cdot_cv,
                                                     w_cv_supp, b_cv,
                                                     col_hidden, partials);
  reduce8_kernel<<<256, 256, 0, stream>>>(partials, v_out);

  // ---- row_next = relu(o_c_out @ W_vc[0:64] + v_out @ W_vc[64:128])
  dense_relu_kernel<<<32, 256, 0, stream>>>(o_c_out, W_vc, v_out, W_vc + 64 * 64,
                                            nullptr, out_row);

  // ---- c -> o : obj2 = relu([obj1|c_o_out]@W_co); bias_ov = obj2 @ W_ov[0:64]
  obj_update_kernel<<<1, 256, 0, stream>>>(out_row, co_supp, obj1,
                                           w_attn_co, b_attn_co, W_co, W_ov,
                                           out_obj, bias_ov);

  dot64_kernel<<<16, 256, 0, stream>>>(out_row, w_vc_row, rowpart, NN);

  // ---- o_v_out = relu(bias_ov + col_hidden @ W_ov[64:128])
  dense_relu_kernel<<<32, 256, 0, stream>>>(col_hidden, W_ov + 64 * 64,
                                            nullptr, nullptr, bias_ov, o_v_out);

  // ---- c -> v : c_out = score_vc @ row_next
  fused_attn_partial_kernel<<<256, 256, 0, stream>>>(vc_supp, vc_supp + PLANE,
                                                     colpart, rowpart,
                                                     w_vc_supp, b_vc,
                                                     out_row, partials);
  reduce8_kernel<<<256, 256, 0, stream>>>(partials, c_out);

  // ---- col_next = relu(o_v_out @ W_cv[0:64] + c_out @ W_cv[64:128])
  dense_relu_kernel<<<32, 256, 0, stream>>>(o_v_out, W_cv, c_out, W_cv + 64 * 64,
                                            nullptr, out_col);

  // ---- passthrough tensors (pure bandwidth; async d2d copies)
  hipMemcpyAsync(out_cv, cv_supp, 2 * PLANE * sizeof(float),
                 hipMemcpyDeviceToDevice, stream);
  hipMemcpyAsync(out_vc, vc_supp, 2 * PLANE * sizeof(float),
                 hipMemcpyDeviceToDevice, stream);
  hipMemcpyAsync(out_vo, vo_supp, 8192 * sizeof(float),
                 hipMemcpyDeviceToDevice, stream);
  hipMemcpyAsync(out_co, co_supp, 8192 * sizeof(float),
                 hipMemcpyDeviceToDevice, stream);
}
